// Classifier_47974784696844
// MI455X (gfx1250) — matrix operand order repaired and checked by reading, not compiled, not run on hardware
//
#include <hip/hip_runtime.h>
#include <hip/hip_bf16.h>

typedef _Float16 v16h __attribute__((ext_vector_type(16)));
typedef _Float16 h8   __attribute__((ext_vector_type(8)));
typedef _Float16 h2   __attribute__((ext_vector_type(2)));
typedef float    v8f  __attribute__((ext_vector_type(8)));

#define D_FEAT     128
#define HIDDEN     256
#define LDS_STRIDE 264   // 256 halfs + 8 halfs pad -> 528B rows, conflict-free B reads
#define LDS_BYTES  (HIDDEN * LDS_STRIDE * 2)

union h16u { v16h v; h8 h[2]; };

// Single-instruction ReLU: v_med3_f32(x, 0, +inf) — no canonicalize max pair.
__device__ __forceinline__ float relu1(float x) {
    return __builtin_amdgcn_fmed3f(x, 0.0f, __builtin_inff());
}

// Build eight 16x32 f16 A fragments for one 16-edge tile.
// ISA A layout (16-bit, 16x32): lanes 0-15 hold M=lane, K = {k*32+0..7, k*32+16..23};
// lanes 16-31 hold the same M with K offset by +8.
__device__ __forceinline__ void build_a_frags(const float* __restrict__ dis,
                                              const float* __restrict__ drg,
                                              int hi, v16h A[8])
{
#pragma unroll
    for (int k = 0; k < 8; ++k) {
        const int kb = k * 32 + hi * 8;       // feature index base (0..255)
        const float* s = (k < 4) ? (dis + kb) : (drg + (kb - D_FEAT));
        float4 f0 = *(const float4*)(s + 0);
        float4 f1 = *(const float4*)(s + 4);
        float4 f2 = *(const float4*)(s + 16);
        float4 f3 = *(const float4*)(s + 20);
        v16h a;
        a[0]=(_Float16)f0.x;  a[1]=(_Float16)f0.y;  a[2]=(_Float16)f0.z;  a[3]=(_Float16)f0.w;
        a[4]=(_Float16)f1.x;  a[5]=(_Float16)f1.y;  a[6]=(_Float16)f1.z;  a[7]=(_Float16)f1.w;
        a[8]=(_Float16)f2.x;  a[9]=(_Float16)f2.y;  a[10]=(_Float16)f2.z; a[11]=(_Float16)f2.w;
        a[12]=(_Float16)f3.x; a[13]=(_Float16)f3.y; a[14]=(_Float16)f3.z; a[15]=(_Float16)f3.w;
        A[k] = a;
    }
}

__global__ __launch_bounds__(256)
void edge_mlp_wmma(const float* __restrict__ xdis,
                   const float* __restrict__ xdrg,
                   const long long* __restrict__ eidx,
                   const float* __restrict__ W1,
                   const float* __restrict__ b1,
                   const float* __restrict__ W2,
                   const float* __restrict__ b2,
                   float* __restrict__ out,
                   long long numEdges)
{
    extern __shared__ _Float16 w1s[];

    // ---- Stage W1 (fp32 -> fp16) into padded LDS, once per workgroup ----
    for (int e2 = threadIdx.x; e2 < HIDDEN * HIDDEN / 2; e2 += blockDim.x) {
        const int n = e2 >> 7;            // row (hidden unit)
        const int k = (e2 & 127) * 2;     // col (input feature), pairs
        const float f0 = W1[n * HIDDEN + k];
        const float f1 = W1[n * HIDDEN + k + 1];
        h2 h; h[0] = (_Float16)f0; h[1] = (_Float16)f1;
        *(h2*)(w1s + n * LDS_STRIDE + k) = h;
    }
    __syncthreads();

    const int lane = threadIdx.x & 31;
    const int m    = lane & 15;           // edge row (A) / hidden col (B,C)
    const int hi   = lane >> 4;           // lane half selects K sub-range

    const int wavesPerBlock = blockDim.x >> 5;
    const int gwave  = blockIdx.x * wavesPerBlock + (threadIdx.x >> 5);
    const int nwaves = gridDim.x * wavesPerBlock;

    const long long pairs = (numEdges + 31) / 32;   // 32 edges (2 M-tiles) per wave iter
    const long long lastE = numEdges - 1;
    const float b2v = b2[0];

    for (long long pair = gwave; pair < pairs; pair += nwaves) {
        const long long e0 = pair * 32;

        // ---- Gather + convert: two 16x256 feature tiles -> 2x8 A fragments ----
        long long eA = e0 + m;       if (eA > lastE) eA = lastE;
        long long eB = e0 + 16 + m;  if (eB > lastE) eB = lastE;
        const float* dis0 = xdis + (size_t)eidx[eA] * D_FEAT;
        const float* drg0 = xdrg + (size_t)eidx[numEdges + eA] * D_FEAT;
        const float* dis1 = xdis + (size_t)eidx[eB] * D_FEAT;
        const float* drg1 = xdrg + (size_t)eidx[numEdges + eB] * D_FEAT;

        v16h A0[8], A1[8];
        build_a_frags(dis0, drg0, hi, A0);
        build_a_frags(dis1, drg1, hi, A1);

        // ---- Prefetch NEXT pair's feature rows while this pair computes ----
        // 4 rows x 512B; (m, hi) lanes cover each row at 128B granularity, the
        // +256B line folds into the instruction's immediate offset.
        const long long nextPair = pair + nwaves;
        if (nextPair < pairs) {
            long long fA = nextPair * 32 + m;  if (fA > lastE) fA = lastE;
            long long fB = fA + 16;            if (fB > lastE) fB = lastE;
            const size_t ho = (size_t)(hi * 128);
            const char* pd0 = (const char*)(xdis + (size_t)eidx[fA] * D_FEAT) + ho;
            const char* pg0 = (const char*)(xdrg + (size_t)eidx[numEdges + fA] * D_FEAT) + ho;
            const char* pd1 = (const char*)(xdis + (size_t)eidx[fB] * D_FEAT) + ho;
            const char* pg1 = (const char*)(xdrg + (size_t)eidx[numEdges + fB] * D_FEAT) + ho;
            __builtin_prefetch(pd0, 0, 3);  __builtin_prefetch(pd0 + 256, 0, 3);
            __builtin_prefetch(pg0, 0, 3);  __builtin_prefetch(pg0 + 256, 0, 3);
            __builtin_prefetch(pd1, 0, 3);  __builtin_prefetch(pd1 + 256, 0, 3);
            __builtin_prefetch(pg1, 0, 3);  __builtin_prefetch(pg1 + 256, 0, 3);
        }

        float part0[8], part1[8];
#pragma unroll
        for (int v = 0; v < 8; ++v) { part0[v] = 0.f; part1[v] = 0.f; }

        // ---- Layer 1: 16 N-tiles x 8 K-steps; every B fragment feeds TWO wmmas.
        //      C starts as inline 0; b1 is folded into the fused epilogue. ----
        for (int j = 0; j < 16; ++j) {
            const int n = j * 16 + m;
            const float b1v = b1[n];
            const float w2v = W2[n];
            v8f c0 = {};   // inline-0 C operand on the first wmma of each chain
            v8f c1 = {};
            // B layout: lane holds column n = j*16+m, K = {k*32 + hi*8 + 0..7, k*32 + 16 + hi*8 + 0..7}
            const _Float16* wrow = w1s + n * LDS_STRIDE + hi * 8;
#pragma unroll
            for (int k = 0; k < 8; ++k) {
                h16u bu;
                bu.h[0] = *(const h8*)(wrow + k * 32);
                bu.h[1] = *(const h8*)(wrow + k * 32 + 16);
                c0 = __builtin_amdgcn_wmma_f32_16x16x32_f16(
                         false, A0[k], false, bu.v, (short)0, c0, false, false);
                c1 = __builtin_amdgcn_wmma_f32_16x16x32_f16(
                         false, A1[k], false, bu.v, (short)0, c1, false, false);
            }
            // ---- Fused layer 2: out[m] += relu(h + b1) * w2 (per-lane partials) ----
#pragma unroll
            for (int v = 0; v < 8; ++v) {
                part0[v] += relu1(c0[v] + b1v) * w2v;
                part1[v] += relu1(c1[v] + b1v) * w2v;
            }
        }

        // ---- Reduce across the 16 lanes of each half-wave (N dimension) ----
#pragma unroll
        for (int v = 0; v < 8; ++v) {
            float s0 = part0[v], s1 = part1[v];
            s0 += __shfl_xor(s0, 8, 32);  s1 += __shfl_xor(s1, 8, 32);
            s0 += __shfl_xor(s0, 4, 32);  s1 += __shfl_xor(s1, 4, 32);
            s0 += __shfl_xor(s0, 2, 32);  s1 += __shfl_xor(s1, 2, 32);
            s0 += __shfl_xor(s0, 1, 32);  s1 += __shfl_xor(s1, 1, 32);
            part0[v] = s0; part1[v] = s1;
        }
        // lanes 0-15 hold rows M=0..7 (VGPR v -> M=v); lanes 16-31 rows 8..15
        if (lane == 0) {
#pragma unroll
            for (int v = 0; v < 8; ++v) {
                if (e0 + v      < numEdges) out[e0 + v]      = part0[v] + b2v;
                if (e0 + 16 + v < numEdges) out[e0 + 16 + v] = part1[v] + b2v;
            }
        } else if (lane == 16) {
#pragma unroll
            for (int v = 0; v < 8; ++v) {
                if (e0 + 8 + v  < numEdges) out[e0 + 8 + v]  = part0[v] + b2v;
                if (e0 + 24 + v < numEdges) out[e0 + 24 + v] = part1[v] + b2v;
            }
        }
    }
}

extern "C" void kernel_launch(void* const* d_in, const int* in_sizes, int n_in,
                              void* d_out, int out_size, void* d_ws, size_t ws_size,
                              hipStream_t stream) {
    const float*      xdis = (const float*)d_in[0];
    const float*      xdrg = (const float*)d_in[1];
    const long long*  eidx = (const long long*)d_in[2];   // int64 indices, [2, E]
    const float*      W1   = (const float*)d_in[3];       // [256, 256]
    const float*      b1   = (const float*)d_in[4];       // [256]
    const float*      W2   = (const float*)d_in[5];       // [1, 256]
    const float*      b2   = (const float*)d_in[6];       // [1]
    float*            outp = (float*)d_out;

    const long long numEdges = (long long)in_sizes[2] / 2;

    // Allow >64KB dynamic LDS (132KB W1 staging; WGP has 320KB).
    (void)hipFuncSetAttribute((const void*)edge_mlp_wmma,
                              hipFuncAttributeMaxDynamicSharedMemorySize,
                              LDS_BYTES);

    dim3 grid(1024), block(256);
    edge_mlp_wmma<<<grid, block, LDS_BYTES, stream>>>(
        xdis, xdrg, eidx, W1, b1, W2, b2, outp, numEdges);
}
